// GraphGenerator_89970974917056
// MI455X (gfx1250) — matrix operand order repaired and checked by reading, not compiled, not run on hardware
//
#include <hip/hip_runtime.h>
#include <hip/hip_bf16.h>
#include <math.h>

// ---------------------------------------------------------------------------
// Types for CDNA5 WMMA (wave32, 16x16x32 bf16 -> f32)
// ---------------------------------------------------------------------------
typedef __bf16 bf16_t;
typedef __attribute__((ext_vector_type(16))) __bf16 v16bf;
typedef __attribute__((ext_vector_type(8)))  __bf16 v8bf;
typedef __attribute__((ext_vector_type(8)))  float  v8f;
typedef __attribute__((ext_vector_type(4)))  int    v4i;

typedef __attribute__((address_space(1))) v4i g_v4i;   // global 128-bit chunk
typedef __attribute__((address_space(3))) v4i l_v4i;   // LDS    128-bit chunk

#if defined(__has_builtin)
#if __has_builtin(__builtin_amdgcn_global_load_async_to_lds_b128)
#define USE_ASYNC_LDS 1
#if __has_builtin(__builtin_amdgcn_s_wait_asynccnt)
#define WAIT_ASYNC() __builtin_amdgcn_s_wait_asynccnt(0)
#else
#define WAIT_ASYNC() asm volatile("s_wait_asynccnt 0" ::: "memory")
#endif
#endif
#endif

static constexpr int N_GRAPH = 100000;
static constexpr int N_CAND  = 10;
static constexpr int N_TOT   = N_GRAPH + N_CAND;   // 100010
static constexpr int F_IN    = 128;
static constexpr int T       = 10;                 // N_TYPES
static constexpr int LDS_PAD = 8;                  // bf16 elems (+16B) per LDS row

// ---------------------------------------------------------------------------
// Small helpers
// ---------------------------------------------------------------------------
__device__ __forceinline__ unsigned f32_ord(float f) {
    unsigned u = __float_as_uint(f);
    return (u & 0x80000000u) ? ~u : (u | 0x80000000u);
}
__device__ __forceinline__ float ord_f32(unsigned u) {
    u = (u & 0x80000000u) ? (u & 0x7fffffffu) : ~u;
    return __uint_as_float(u);
}
__device__ __forceinline__ unsigned hash_u32(unsigned a) {
    a ^= a >> 16; a *= 0x7feb352du;
    a ^= a >> 15; a *= 0x846ca68bu;
    a ^= a >> 16; return a;
}
__device__ __forceinline__ float gumbel_from(unsigned seed, unsigned idx) {
    unsigned h = hash_u32(seed ^ hash_u32(idx));
    float u = ((float)h + 0.5f) * 2.3283064365386963e-10f; // (0,1)
    return -logf(-logf(u));
}

// ---------------------------------------------------------------------------
// Fills
// ---------------------------------------------------------------------------
__global__ void k_fillf(float* p, size_t cnt, float v) {
    size_t i = (size_t)blockIdx.x * blockDim.x + threadIdx.x;
    if (i < cnt) p[i] = v;
}
__global__ void k_fillu(unsigned* p, size_t cnt, unsigned v) {
    size_t i = (size_t)blockIdx.x * blockDim.x + threadIdx.x;
    if (i < cnt) p[i] = v;
}

// ---------------------------------------------------------------------------
// Build H0 bf16 [n, 128] = concat(x, candidate_set)
// ---------------------------------------------------------------------------
__global__ void k_build_h0(const float* __restrict__ x, const float* __restrict__ cand,
                           bf16_t* __restrict__ H0) {
    size_t i = (size_t)blockIdx.x * blockDim.x + threadIdx.x;
    if (i >= (size_t)N_TOT * F_IN) return;
    int row = (int)(i >> 7), col = (int)(i & 127);
    float v = (row < N_GRAPH) ? x[(size_t)row * F_IN + col]
                              : cand[(size_t)(row - N_GRAPH) * F_IN + col];
    H0[i] = (bf16_t)v;
}

// ---------------------------------------------------------------------------
// Weight prep: W f32 [K, F] row-major -> Wt bf16 [Fpad, Kpad] (transposed, 0-pad)
// ---------------------------------------------------------------------------
__global__ void k_prep_w(const float* __restrict__ W, int K, int F, int Kpad, int Fpad,
                         bf16_t* __restrict__ Wt) {
    int i = blockIdx.x * blockDim.x + threadIdx.x;
    if (i >= Fpad * Kpad) return;
    int nn = i / Kpad, kk = i % Kpad;
    float v = (nn < F && kk < K) ? W[(size_t)kk * F + nn] : 0.0f;
    Wt[i] = (bf16_t)v;
}

// ---------------------------------------------------------------------------
// Edge preprocessing: int64 -> int32 src/dst (+ degree), then per-edge coef
// ---------------------------------------------------------------------------
__global__ void k_edge_prep(const long long* __restrict__ ei, int E,
                            int* __restrict__ src, int* __restrict__ dst,
                            float* __restrict__ deg) {
    int e = blockIdx.x * blockDim.x + threadIdx.x;
    if (e >= E) return;
    int s = (int)ei[e];
    int d = (int)ei[(size_t)E + e];
    src[e] = s; dst[e] = d;
    atomicAdd(&deg[d], 1.0f);
}
__global__ void k_dinv(const float* __restrict__ deg, float* __restrict__ dinv, int n) {
    int i = blockIdx.x * blockDim.x + threadIdx.x;
    if (i < n) dinv[i] = rsqrtf(deg[i]);
}
__global__ void k_edge_coef(const int* __restrict__ src, const int* __restrict__ dst,
                            const float* __restrict__ dinv, float* __restrict__ coef, int E) {
    int e = blockIdx.x * blockDim.x + threadIdx.x;
    if (e >= E) return;
    coef[e] = dinv[src[e]] * dinv[dst[e]];
}

// ---------------------------------------------------------------------------
// WMMA GEMM: out[rows, Fpad] = A[rows, Kpad](bf16) @ Wt[Fpad, Kpad]^T (+bias, relu6)
// One wave -> one 16x16 f32 tile. Weight tile staged in LDS (async if available).
// ---------------------------------------------------------------------------
__global__ void k_gemm(const bf16_t* __restrict__ A, int Kpad,
                       const bf16_t* __restrict__ Wt,
                       const float* __restrict__ bias,
                       float* __restrict__ outF, bf16_t* __restrict__ outB,
                       int Fpad, int F, int rows, int act_relu6) {
    __shared__ bf16_t sW[32 * (128 + LDS_PAD)];   // max tile 32x128 bf16 (+row pad)
    const int ldsw = Kpad + LDS_PAD;
    const int cpr  = Kpad >> 3;                   // 16B chunks per weight row
    const int chunks = Fpad * cpr;
    for (int c = threadIdx.x; c < chunks; c += blockDim.x) {
        int rw = c / cpr, co = (c - rw * cpr) * 8;
#ifdef USE_ASYNC_LDS
        __builtin_amdgcn_global_load_async_to_lds_b128(
            (g_v4i*)(Wt + (size_t)rw * Kpad + co),
            (l_v4i*)(sW + rw * ldsw + co), 0, 0);
#else
        *reinterpret_cast<v8bf*>(sW + rw * ldsw + co) =
            *reinterpret_cast<const v8bf*>(Wt + (size_t)rw * Kpad + co);
#endif
    }
#ifdef USE_ASYNC_LDS
    WAIT_ASYNC();
#endif
    __syncthreads();

    const int lane = threadIdx.x & 31;
    const int wid  = (blockIdx.x * blockDim.x + threadIdx.x) >> 5;
    const int tilesN = Fpad >> 4;
    const int tm = wid / tilesN;
    const int tn = wid % tilesN;
    const int g  = lane >> 4;            // half-wave group
    const int hl = lane & 15;

    int rowA = tm * 16 + hl;
    if (rowA >= rows) rowA = rows - 1;   // clamp reads; stores masked below
    const bf16_t* arow = A + (size_t)rowA * Kpad;
    const bf16_t* brow = sW + (tn * 16 + hl) * ldsw;

    v8f c = {};
    for (int k0 = 0; k0 < Kpad; k0 += 32) {
        // A 16-bit fragment: elems 0..7 -> K=k0+g*8.., elems 8..15 -> K=k0+16+g*8..
        v8bf alo = *reinterpret_cast<const v8bf*>(arow + k0 + g * 8);
        v8bf ahi = *reinterpret_cast<const v8bf*>(arow + k0 + 16 + g * 8);
        // B 16-bit fragment: lane holds N=hl, elems 0..7 -> K=k0+g*8.., elems 8..15 -> K=k0+16+g*8..
        v8bf blo = *reinterpret_cast<const v8bf*>(brow + k0 + g * 8);
        v8bf bhi = *reinterpret_cast<const v8bf*>(brow + k0 + 16 + g * 8);
        v16bf a, b;
#pragma unroll
        for (int j = 0; j < 8; ++j) {
            a[j] = alo[j]; a[8 + j] = ahi[j];
            b[j] = blo[j]; b[8 + j] = bhi[j];
        }
        c = __builtin_amdgcn_wmma_f32_16x16x32_bf16(false, a, false, b,
                                                    (short)0, c, false, false);
    }

    const int N = tn * 16 + hl;
#pragma unroll
    for (int r = 0; r < 8; ++r) {
        int M = tm * 16 + r + g * 8;     // C/D layout: VGPR r -> M = r + g*8
        if (M < rows) {
            float v = c[r];
            if (bias && N < F) v += bias[N];
            if (N >= F) v = 0.0f;        // keep padding columns exactly zero
            if (act_relu6) v = fminf(fmaxf(v, 0.0f), 6.0f);
            outF[(size_t)M * Fpad + N] = v;
            if (outB) outB[(size_t)M * Fpad + N] = (bf16_t)v;
        }
    }
}

// ---------------------------------------------------------------------------
// Edge aggregation: agg[dst] += hW[src] * coef   (Fpad fixed = 32)
// ---------------------------------------------------------------------------
__global__ void k_edge_agg(const int* __restrict__ src, const int* __restrict__ dst,
                           const float* __restrict__ coef, int E,
                           const float* __restrict__ hW, float* __restrict__ agg,
                           int Fq /* F/4 */) {
    int e = blockIdx.x * blockDim.x + threadIdx.x;
    if (e >= E) return;
    int s = src[e], d = dst[e];
    float cf = coef[e];
    const float4* hs = reinterpret_cast<const float4*>(hW + (size_t)s * 32);
    float*        ad = agg + (size_t)d * 32;
    __builtin_prefetch(ad, 1, 1);   // pull atomic target lines toward L2
    for (int q = 0; q < Fq; ++q) {
        float4 v = hs[q];
        atomicAdd(&ad[4 * q + 0], v.x * cf);
        atomicAdd(&ad[4 * q + 1], v.y * cf);
        atomicAdd(&ad[4 * q + 2], v.z * cf);
        atomicAdd(&ad[4 * q + 3], v.w * cf);
    }
}

// ---------------------------------------------------------------------------
// GCN finalize: h = agg + hW*dinv^2 + b  -> bf16 (padded cols stay 0)
// ---------------------------------------------------------------------------
__global__ void k_gcn_fin(const float* __restrict__ agg, const float* __restrict__ hW,
                          const float* __restrict__ dinv, const float* __restrict__ b,
                          bf16_t* __restrict__ hB, int n, int Fpad, int F) {
    size_t i = (size_t)blockIdx.x * blockDim.x + threadIdx.x;
    if (i >= (size_t)n * Fpad) return;
    int row = (int)(i / Fpad), f = (int)(i % Fpad);
    float di = dinv[row];
    float v = (f < F) ? (agg[i] + hW[i] * di * di + b[f]) : 0.0f;
    hB[i] = (bf16_t)v;
}

// ---------------------------------------------------------------------------
// Column-wise (axis=0) softmax reductions over logits [rows, Fpad] (T=10 cols)
// ---------------------------------------------------------------------------
__global__ void k_colmax(const float* __restrict__ logits, int rows, int Fpad,
                         unsigned* __restrict__ cmax) {
    int tid = blockIdx.x * blockDim.x + threadIdx.x;
    int nth = gridDim.x * blockDim.x;
    float m[T];
#pragma unroll
    for (int t = 0; t < T; ++t) m[t] = -3.4e38f;
    for (int r = tid; r < rows; r += nth)
#pragma unroll
        for (int t = 0; t < T; ++t) m[t] = fmaxf(m[t], logits[(size_t)r * Fpad + t]);
#pragma unroll
    for (int t = 0; t < T; ++t) atomicMax(&cmax[t], f32_ord(m[t]));
}

__global__ void k_colsum(const float* __restrict__ logits, int rows, int Fpad,
                         const unsigned* __restrict__ cmax, float* __restrict__ csum) {
    int tid = blockIdx.x * blockDim.x + threadIdx.x;
    int nth = gridDim.x * blockDim.x;
    float m[T], s[T];
#pragma unroll
    for (int t = 0; t < T; ++t) { m[t] = ord_f32(cmax[t]); s[t] = 0.0f; }
    for (int r = tid; r < rows; r += nth)
#pragma unroll
        for (int t = 0; t < T; ++t) s[t] += expf(logits[(size_t)r * Fpad + t] - m[t]);
#pragma unroll
    for (int t = 0; t < T; ++t) atomicAdd(&csum[t], s[t]);
}

// ---------------------------------------------------------------------------
// Start head: probs (mask candidates, where==0 -> 1e-10), Gumbel-argmax sample
// ---------------------------------------------------------------------------
__global__ void k_start(const float* __restrict__ logits, int n, int Fpad,
                        const unsigned* __restrict__ cmax, const float* __restrict__ csum,
                        float* __restrict__ out_probs, float* __restrict__ out_node,
                        int* __restrict__ start_idx) {
    int r = blockIdx.x * blockDim.x + threadIdx.x;
    if (r >= n) return;
    float best = -3.4e38f; int bi = 0;
#pragma unroll
    for (int t = 0; t < T; ++t) {
        float p = expf(logits[(size_t)r * Fpad + t] - ord_f32(cmax[t])) / csum[t];
        if (r >= N_GRAPH) p = 0.0f;             // mask candidate rows
        if (p == 0.0f) p = 1e-10f;
        out_probs[(size_t)r * T + t] = p;
        float sc = logf(p) + gumbel_from(0x9E3779B9u, (unsigned)(r * T + t));
        if (sc > best) { best = sc; bi = t; }
    }
    out_node[r]  = (float)bi;
    start_idx[r] = bi;
}

// ---------------------------------------------------------------------------
// combined = concat(h, h[start_node])  (bf16, [2n, 32])
// ---------------------------------------------------------------------------
__global__ void k_combined(const bf16_t* __restrict__ h, const int* __restrict__ start_idx,
                           bf16_t* __restrict__ comb, int n) {
    size_t i = (size_t)blockIdx.x * blockDim.x + threadIdx.x;
    if (i >= (size_t)2 * n * 32) return;
    int row = (int)(i >> 5), col = (int)(i & 31);
    if (row < n) comb[i] = h[i];
    else         comb[i] = h[(size_t)start_idx[row - n] * 32 + col];
}

// ---------------------------------------------------------------------------
// End head: write probs, zero rows indexed by start_node, where+sample
// ---------------------------------------------------------------------------
__global__ void k_end_probs(const float* __restrict__ logits, int rows, int Fpad,
                            const unsigned* __restrict__ cmax, const float* __restrict__ csum,
                            float* __restrict__ out_probs) {
    int r = blockIdx.x * blockDim.x + threadIdx.x;
    if (r >= rows) return;
#pragma unroll
    for (int t = 0; t < T; ++t)
        out_probs[(size_t)r * T + t] =
            expf(logits[(size_t)r * Fpad + t] - ord_f32(cmax[t])) / csum[t];
}
__global__ void k_end_zero(const int* __restrict__ start_idx, int n,
                           float* __restrict__ out_probs) {
    int i = blockIdx.x * blockDim.x + threadIdx.x;
    if (i >= n) return;
    int row = start_idx[i];
#pragma unroll
    for (int t = 0; t < T; ++t) out_probs[(size_t)row * T + t] = 0.0f;
}
__global__ void k_end_sample(float* __restrict__ out_probs, int rows,
                             float* __restrict__ out_node) {
    int r = blockIdx.x * blockDim.x + threadIdx.x;
    if (r >= rows) return;
    float best = -3.4e38f; int bi = 0;
#pragma unroll
    for (int t = 0; t < T; ++t) {
        float p = out_probs[(size_t)r * T + t];
        if (p == 0.0f) { p = 1e-10f; out_probs[(size_t)r * T + t] = p; }
        float sc = logf(p) + gumbel_from(0x85EBCA6Bu, (unsigned)(r * T + t));
        if (sc > best) { best = sc; bi = t; }
    }
    out_node[r] = (float)bi;
}

// ---------------------------------------------------------------------------
// Host orchestration
// ---------------------------------------------------------------------------
static inline int nblk(size_t n, int b) { return (int)((n + (size_t)b - 1) / (size_t)b); }

extern "C" void kernel_launch(void* const* d_in, const int* in_sizes, int n_in,
                              void* d_out, int out_size, void* d_ws, size_t ws_size,
                              hipStream_t stream) {
    const float*     x    = (const float*)d_in[0];
    const float*     cand = (const float*)d_in[1];
    const long long* ei   = (const long long*)d_in[2];
    const float* W1  = (const float*)d_in[4];  const float* b1  = (const float*)d_in[5];
    const float* W2  = (const float*)d_in[6];  const float* b2  = (const float*)d_in[7];
    const float* W3  = (const float*)d_in[8];  const float* b3  = (const float*)d_in[9];
    const float* Ws1 = (const float*)d_in[10]; const float* bs1 = (const float*)d_in[11];
    const float* Ws2 = (const float*)d_in[12]; const float* bs2 = (const float*)d_in[13];
    const float* We1 = (const float*)d_in[14]; const float* be1 = (const float*)d_in[15];
    const float* We2 = (const float*)d_in[16]; const float* be2 = (const float*)d_in[17];

    const int n  = N_TOT;
    const int n2 = 2 * N_TOT;
    const int E  = in_sizes[2] / 2;

    // workspace arena
    char* ws = (char*)d_ws; size_t off = 0;
    auto alloc = [&](size_t bytes) -> void* {
        void* p = ws + off; off = (off + bytes + 255) & ~(size_t)255; return p;
    };
    float*    deg     = (float*)alloc((size_t)n * 4);
    float*    dinv    = (float*)alloc((size_t)n * 4);
    int*      src32   = (int*)alloc((size_t)E * 4);
    int*      dst32   = (int*)alloc((size_t)E * 4);
    float*    coef    = (float*)alloc((size_t)E * 4);
    bf16_t*   H0      = (bf16_t*)alloc((size_t)n * F_IN * 2);
    float*    bufA    = (float*)alloc((size_t)n2 * 32 * 4);   // hW / s1F / e1F
    float*    agg     = (float*)alloc((size_t)n * 32 * 4);
    bf16_t*   hB      = (bf16_t*)alloc((size_t)n * 32 * 2);
    bf16_t*   s1B     = (bf16_t*)alloc((size_t)n * 32 * 2);
    bf16_t*   comb    = (bf16_t*)alloc((size_t)n2 * 32 * 2);
    bf16_t*   e1B     = (bf16_t*)alloc((size_t)n2 * 32 * 2);
    float*    logitsS = (float*)alloc((size_t)n * 16 * 4);
    float*    logitsE = (float*)alloc((size_t)n2 * 16 * 4);
    bf16_t*   W1t  = (bf16_t*)alloc(32 * 128 * 2);
    bf16_t*   W2t  = (bf16_t*)alloc(32 * 32 * 2);
    bf16_t*   W3t  = (bf16_t*)alloc(32 * 32 * 2);
    bf16_t*   Ws1t = (bf16_t*)alloc(32 * 32 * 2);
    bf16_t*   Ws2t = (bf16_t*)alloc(16 * 32 * 2);
    bf16_t*   We1t = (bf16_t*)alloc(32 * 32 * 2);
    bf16_t*   We2t = (bf16_t*)alloc(16 * 32 * 2);
    unsigned* cmax = (unsigned*)alloc(T * 4);
    float*    csum = (float*)alloc(T * 4);
    int*      sidx = (int*)alloc((size_t)n * 4);

    float* out           = (float*)d_out;
    float* out_startnode = out;                   // [n]
    float* out_endnode   = out + n;               // [2n]
    float* out_startp    = out + 3 * (size_t)n;   // [n, T]
    float* out_endp      = out + 13 * (size_t)n;  // [2n, T]

    const int B = 256;
    auto gemm = [&](const bf16_t* A, int Kpad, const bf16_t* Wt, const float* bias,
                    float* oF, bf16_t* oB, int Fpad, int F, int rows, int act) {
        long long waves = (long long)((rows + 15) / 16) * (Fpad / 16);
        k_gemm<<<nblk((size_t)waves * 32, B), B, 0, stream>>>(A, Kpad, Wt, bias, oF, oB,
                                                             Fpad, F, rows, act);
    };

    // ---- prep -------------------------------------------------------------
    k_build_h0<<<nblk((size_t)n * F_IN, B), B, 0, stream>>>(x, cand, H0);
    k_prep_w<<<nblk(32 * 128, B), B, 0, stream>>>(W1, 128, 16, 128, 32, W1t);
    k_prep_w<<<nblk(32 * 32, B), B, 0, stream>>>(W2, 16, 24, 32, 32, W2t);
    k_prep_w<<<nblk(32 * 32, B), B, 0, stream>>>(W3, 24, 32, 32, 32, W3t);
    k_prep_w<<<nblk(32 * 32, B), B, 0, stream>>>(Ws1, 32, 16, 32, 32, Ws1t);
    k_prep_w<<<nblk(16 * 32, B), B, 0, stream>>>(Ws2, 16, 10, 32, 16, Ws2t);
    k_prep_w<<<nblk(32 * 32, B), B, 0, stream>>>(We1, 32, 24, 32, 32, We1t);
    k_prep_w<<<nblk(16 * 32, B), B, 0, stream>>>(We2, 24, 10, 32, 16, We2t);

    k_fillf<<<nblk(n, B), B, 0, stream>>>(deg, n, 1.0f);     // +1 (self-loop)
    k_edge_prep<<<nblk(E, B), B, 0, stream>>>(ei, E, src32, dst32, deg);
    k_dinv<<<nblk(n, B), B, 0, stream>>>(deg, dinv, n);
    k_edge_coef<<<nblk(E, B), B, 0, stream>>>(src32, dst32, dinv, coef, E);

    // ---- 3 GCN layers -----------------------------------------------------
    const bf16_t* Win = H0; int Kp = 128;
    const bf16_t* Wts[3] = { W1t, W2t, W3t };
    const float*  bs[3]  = { b1, b2, b3 };
    const int     Fs[3]  = { 16, 24, 32 };
    for (int l = 0; l < 3; ++l) {
        gemm(Win, Kp, Wts[l], nullptr, bufA, nullptr, 32, Fs[l], n, 0);
        k_fillf<<<nblk((size_t)n * 32, B), B, 0, stream>>>(agg, (size_t)n * 32, 0.0f);
        k_edge_agg<<<nblk(E, B), B, 0, stream>>>(src32, dst32, coef, E, bufA, agg,
                                                 Fs[l] / 4);
        k_gcn_fin<<<nblk((size_t)n * 32, B), B, 0, stream>>>(agg, bufA, dinv, bs[l],
                                                             hB, n, 32, Fs[l]);
        Win = hB; Kp = 32;
    }

    // ---- start head -------------------------------------------------------
    gemm(hB, 32, Ws1t, bs1, bufA, s1B, 32, 16, n, 1);             // relu6
    gemm(s1B, 32, Ws2t, bs2, logitsS, nullptr, 16, 10, n, 0);
    k_fillu<<<1, 32, 0, stream>>>(cmax, T, 0u);
    k_fillf<<<1, 32, 0, stream>>>(csum, T, 0.0f);
    k_colmax<<<256, B, 0, stream>>>(logitsS, n, 16, cmax);
    k_colsum<<<256, B, 0, stream>>>(logitsS, n, 16, cmax, csum);
    k_start<<<nblk(n, B), B, 0, stream>>>(logitsS, n, 16, cmax, csum,
                                          out_startp, out_startnode, sidx);

    // ---- end head ---------------------------------------------------------
    k_combined<<<nblk((size_t)n2 * 32, B), B, 0, stream>>>(hB, sidx, comb, n);
    gemm(comb, 32, We1t, be1, bufA, e1B, 32, 24, n2, 1);          // relu6
    gemm(e1B, 32, We2t, be2, logitsE, nullptr, 16, 10, n2, 0);
    k_fillu<<<1, 32, 0, stream>>>(cmax, T, 0u);
    k_fillf<<<1, 32, 0, stream>>>(csum, T, 0.0f);
    k_colmax<<<256, B, 0, stream>>>(logitsE, n2, 16, cmax);
    k_colsum<<<256, B, 0, stream>>>(logitsE, n2, 16, cmax, csum);
    k_end_probs<<<nblk(n2, B), B, 0, stream>>>(logitsE, n2, 16, cmax, csum, out_endp);
    k_end_zero<<<nblk(n, B), B, 0, stream>>>(sidx, n, out_endp);
    k_end_sample<<<nblk(n2, B), B, 0, stream>>>(out_endp, n2, out_endnode);

    (void)n_in; (void)out_size; (void)ws_size;
}
